// SpatialAttentionLayer_60378650247585
// MI455X (gfx1250) — matrix operand order repaired and checked by reading, not compiled, not run on hardware
//
#include <hip/hip_runtime.h>
#include <hip/hip_bf16.h>

typedef __attribute__((ext_vector_type(16))) _Float16 v16h;
typedef __attribute__((ext_vector_type(8)))  _Float16 v8h;
typedef __attribute__((ext_vector_type(8)))  float    v8f;
typedef __attribute__((ext_vector_type(4)))  float    v4f;

#define HDIM  128
#define NQ    4096
#define NR    4096
#define BATCH 8

// ---------------------------------------------------------------------------
// D = A(16x32 f16) x B(32x16 f16) + C(16x16 f32), wave32 CDNA5 WMMA
// ---------------------------------------------------------------------------
__device__ __forceinline__ v8f wmma_16x16x32(v16h a, v16h b, v8f c) {
  return __builtin_amdgcn_wmma_f32_16x16x32_f16(
      /*neg_a=*/false, a, /*neg_b=*/false, b,
      /*c_mod=*/(short)0, c, /*reuse_a=*/false, /*reuse_b=*/false);
}

// A-fragment (16x32) from row-major f32 matrix, row fixed per lane.
// ISA layout: half i -> K = i + 8*(i>=8) + 8*hi   (hi = lane>=16)
__device__ __forceinline__ v16h load_a_f32(const float* __restrict__ X,
                                           size_t row, int ld, int k0, int hi) {
  const float* p0 = X + row * (size_t)ld + k0 + 8 * hi;       // K = k0+8hi+0..7
  const float* p1 = X + row * (size_t)ld + k0 + 16 + 8 * hi;  // K = k0+16+8hi+0..7
  v4f a0 = *(const v4f*)p0;
  v4f a1 = *(const v4f*)(p0 + 4);
  v4f a2 = *(const v4f*)p1;
  v4f a3 = *(const v4f*)(p1 + 4);
  v16h a;
#pragma unroll
  for (int i = 0; i < 4; ++i) {
    a[i]      = (_Float16)a0[i];
    a[i + 4]  = (_Float16)a1[i];
    a[i + 8]  = (_Float16)a2[i];
    a[i + 12] = (_Float16)a3[i];
  }
  return a;
}

// B-fragment (32x16) from K-contiguous f16 matrix T[n][k] (row stride ldk).
// ISA layout: lane n = lane&15, half i -> K = i + 8*(i>=8) + 8*hi.
__device__ __forceinline__ v16h load_b_f16(const _Float16* __restrict__ T,
                                           int n, int ldk, int k0, int hi) {
  const _Float16* p = T + (size_t)n * ldk + k0 + 8 * hi;
  v8h lo = *(const v8h*)p;
  v8h hh = *(const v8h*)(p + 16);
  return __builtin_shufflevector(lo, hh, 0, 1, 2, 3, 4, 5, 6, 7,
                                 8, 9, 10, 11, 12, 13, 14, 15);
}

// A-fragment (16x32) from f16 LDS tile, row-major, row fixed per lane.
__device__ __forceinline__ v16h load_a_lds(const _Float16* __restrict__ T,
                                           int row, int ld, int k0, int hi) {
  const _Float16* p0 = T + row * ld + k0 + 8 * hi;
  const _Float16* p1 = T + row * ld + k0 + 16 + 8 * hi;
  v8h lo = *(const v8h*)p0;
  v8h hh = *(const v8h*)p1;
  return __builtin_shufflevector(lo, hh, 0, 1, 2, 3, 4, 5, 6, 7,
                                 8, 9, 10, 11, 12, 13, 14, 15);
}

// ---------------------------------------------------------------------------
// Transpose + f32->f16 convert a (K,N) weight into Wt[n][k]
// ---------------------------------------------------------------------------
__global__ void transpose_f16_kernel(const float* __restrict__ W,
                                     _Float16* __restrict__ Wt, int K, int N) {
  int idx = blockIdx.x * blockDim.x + threadIdx.x;
  if (idx < K * N) {
    int k = idx / N, n = idx % N;
    Wt[(size_t)n * K + k] = (_Float16)W[idx];
  }
}

// ---------------------------------------------------------------------------
// pr (B,NR,H) f32  ->  prT (B,H,NR) f16   (one-shot; L2-resident afterwards)
// Block = 128 threads, one (batch, 32-ref tile). Lane = h: coalesced reads,
// 64B-contiguous vector writes per thread.
// ---------------------------------------------------------------------------
__global__ void __launch_bounds__(128)
convT_kernel(const float* __restrict__ pr, _Float16* __restrict__ prT) {
  const int tid = threadIdx.x;
  const int b = blockIdx.x >> 7;           // NR/32 = 128 tiles per batch
  const int r0 = (blockIdx.x & 127) * 32;
  _Float16 buf[32];
#pragma unroll
  for (int r = 0; r < 32; ++r)
    buf[r] = (_Float16)pr[((size_t)b * NR + r0 + r) * HDIM + tid];
  _Float16* dst = prT + ((size_t)b * HDIM + tid) * NR + r0;
#pragma unroll
  for (int c = 0; c < 4; ++c) {
    v8h v;
#pragma unroll
    for (int j = 0; j < 8; ++j) v[j] = buf[c * 8 + j];
    *(v8h*)(dst + c * 8) = v;
  }
}

// ---------------------------------------------------------------------------
// Two-layer MLP: Y = relu(X @ W1 + b1) @ W2 + b2     (X: rows x 128)
// Block = 128 threads (4 waves), 64 rows per block; WMMA f16 -> f32.
// ---------------------------------------------------------------------------
__global__ void __launch_bounds__(128)
mlp2_kernel(const float* __restrict__ X, const _Float16* __restrict__ Wt1,
            const float* __restrict__ b1, const _Float16* __restrict__ Wt2,
            const float* __restrict__ b2, float* __restrict__ Y) {
  __shared__ __align__(16) _Float16 hid[64 * HDIM];
  const int tid = threadIdx.x;
  const int w = tid >> 5, L = tid & 31, lm = L & 15, hi = L >> 4;
  const size_t row0 = (size_t)blockIdx.x * 64;
  const size_t mrow = row0 + w * 16 + lm;

  v8f acc[8];
#pragma unroll
  for (int nt = 0; nt < 8; ++nt) acc[nt] = {};

#pragma unroll
  for (int k0 = 0; k0 < HDIM; k0 += 32) {
    v16h a = load_a_f32(X, mrow, HDIM, k0, hi);
#pragma unroll
    for (int nt = 0; nt < 8; ++nt) {
      v16h b = load_b_f16(Wt1, nt * 16 + lm, HDIM, k0, hi);
      acc[nt] = wmma_16x16x32(a, b, acc[nt]);
    }
  }
  // bias + ReLU -> f16 LDS (row-major), rows owned by this wave only
#pragma unroll
  for (int nt = 0; nt < 8; ++nt) {
    const int n = nt * 16 + lm;
    const float bias = b1[n];
#pragma unroll
    for (int v = 0; v < 8; ++v) {
      float hv = fmaxf(acc[nt][v] + bias, 0.0f);
      hid[(w * 16 + v + 8 * hi) * HDIM + n] = (_Float16)hv;
    }
  }
  __syncthreads();

  v8f acc2[8];
#pragma unroll
  for (int nt = 0; nt < 8; ++nt) acc2[nt] = {};
  const int arow = w * 16 + lm;
#pragma unroll
  for (int k0 = 0; k0 < HDIM; k0 += 32) {
    v16h a = load_a_lds(hid, arow, HDIM, k0, hi);
#pragma unroll
    for (int nt = 0; nt < 8; ++nt) {
      v16h b = load_b_f16(Wt2, nt * 16 + lm, HDIM, k0, hi);
      acc2[nt] = wmma_16x16x32(a, b, acc2[nt]);
    }
  }
#pragma unroll
  for (int nt = 0; nt < 8; ++nt) {
    const int n = nt * 16 + lm;
    const float bias = b2[n];
#pragma unroll
    for (int v = 0; v < 8; ++v)
      Y[(row0 + w * 16 + v + 8 * hi) * HDIM + n] = acc2[nt][v] + bias;
  }
}

// ---------------------------------------------------------------------------
// Fused attention: generate Gaussian-kernel attention tile, stream it to the
// attention output, and WMMA-accumulate attended = attention @ pr.
// Block = 128 threads, one (batch, 64-query) tile; K-loop over 4096 refs.
// No LDS, no barriers: B-fragments come straight from L2-resident prT,
// ref locations from two contiguous 64B runs (L0-cached, shared by lanes).
// ---------------------------------------------------------------------------
__global__ void __launch_bounds__(128)
attn_kernel(const float* __restrict__ qloc, const float* __restrict__ rloc,
            const _Float16* __restrict__ prT, float* __restrict__ attn_out,
            float* __restrict__ attended) {
  const int tid = threadIdx.x;
  const int w = tid >> 5, L = tid & 31, lm = L & 15, hi = L >> 4;
  const int b = blockIdx.x >> 6;       // NQ/64 = 64 query tiles per batch
  const int qt = blockIdx.x & 63;
  const int q = qt * 64 + w * 16 + lm; // this lane's query row (A-matrix M)

  const float qx = qloc[((size_t)b * NQ + q) * 2 + 0];
  const float qy = qloc[((size_t)b * NQ + q) * 2 + 1];
  float* __restrict__ arow = attn_out + ((size_t)b * NQ + q) * NR;
  const float* __restrict__ rbase = rloc + (size_t)b * NR * 2;
  const _Float16* __restrict__ pbase = prT + (size_t)b * HDIM * NR;

  v8f acc[8];
#pragma unroll
  for (int nt = 0; nt < 8; ++nt) acc[nt] = {};

  for (int r0 = 0; r0 < NR; r0 += 32) {
    // (x,y) pairs for this lane's K values: two contiguous 64B runs
    const float* rp0 = rbase + (size_t)(r0 + 8 * hi) * 2;        // K = 8hi+0..7
    const float* rp1 = rbase + (size_t)(r0 + 16 + 8 * hi) * 2;   // K = 16+8hi+0..7
    v4f f0[4], f1[4];
#pragma unroll
    for (int c = 0; c < 4; ++c) {
      f0[c] = ((const v4f*)rp0)[c];
      f1[c] = ((const v4f*)rp1)[c];
    }
    // 16 attention values per lane, in exact A-fragment order
    float vals[16];
    v16h a;
#pragma unroll
    for (int i = 0; i < 8; ++i) {
      const float dx = f0[i >> 1][(i & 1) * 2] - qx;
      const float dy = f0[i >> 1][(i & 1) * 2 + 1] - qy;
      float lg = -50.0f * (dx * dx + dy * dy);   // -0.5*((d/0.1)^2)
      lg = fminf(fmaxf(lg, -10.0f), 10.0f);
      vals[i] = __expf(lg);
      a[i] = (_Float16)vals[i];
    }
#pragma unroll
    for (int i = 0; i < 8; ++i) {
      const float dx = f1[i >> 1][(i & 1) * 2] - qx;
      const float dy = f1[i >> 1][(i & 1) * 2 + 1] - qy;
      float lg = -50.0f * (dx * dx + dy * dy);
      lg = fminf(fmaxf(lg, -10.0f), 10.0f);
      vals[8 + i] = __expf(lg);
      a[8 + i] = (_Float16)vals[8 + i];
    }
    // stream attention output: two contiguous 8-float runs per lane
    {
      float* p0 = arow + r0 + 8 * hi;
      v4f s0 = {vals[0], vals[1], vals[2], vals[3]};
      v4f s1 = {vals[4], vals[5], vals[6], vals[7]};
      *(v4f*)p0 = s0;
      *(v4f*)(p0 + 4) = s1;
      float* p1 = arow + r0 + 16 + 8 * hi;
      v4f s2 = {vals[8], vals[9], vals[10], vals[11]};
      v4f s3 = {vals[12], vals[13], vals[14], vals[15]};
      *(v4f*)p1 = s2;
      *(v4f*)(p1 + 4) = s3;
    }
    // attended += attention_tile @ pr_tile  (B straight from L2-hot prT)
#pragma unroll
    for (int nt = 0; nt < 8; ++nt) {
      v16h bf = load_b_f16(pbase, nt * 16 + lm, NR, r0, hi);
      acc[nt] = wmma_16x16x32(a, bf, acc[nt]);
    }
  }

#pragma unroll
  for (int nt = 0; nt < 8; ++nt)
#pragma unroll
    for (int v = 0; v < 8; ++v)
      attended[((size_t)b * NQ + qt * 64 + w * 16 + v + 8 * hi) * HDIM +
               nt * 16 + lm] = acc[nt][v];
}

// ---------------------------------------------------------------------------
// Final MLP over concat(pq, attended) (K=256) + LayerNorm.
// ---------------------------------------------------------------------------
__global__ void __launch_bounds__(128)
final_kernel(const float* __restrict__ pq, const float* __restrict__ att,
             const _Float16* __restrict__ Wtf1, const float* __restrict__ bf1,
             const _Float16* __restrict__ Wtf2, const float* __restrict__ bf2,
             const float* __restrict__ ln_g, const float* __restrict__ ln_b,
             float* __restrict__ out) {
  __shared__ __align__(16) _Float16 hid[64 * HDIM];
  __shared__ float fin[64 * HDIM];
  __shared__ float mu_s[64], rs_s[64];
  const int tid = threadIdx.x;
  const int w = tid >> 5, L = tid & 31, lm = L & 15, hi = L >> 4;
  const size_t row0 = (size_t)blockIdx.x * 64;
  const size_t mrow = row0 + w * 16 + lm;

  v8f acc[8];
#pragma unroll
  for (int nt = 0; nt < 8; ++nt) acc[nt] = {};

  // layer 1: K = 256 (first 128 from pq, next 128 from attended)
#pragma unroll
  for (int k0 = 0; k0 < 256; k0 += 32) {
    const float* src = (k0 < 128) ? pq : att;
    const int kk = (k0 < 128) ? k0 : (k0 - 128);
    v16h a = load_a_f32(src, mrow, HDIM, kk, hi);
#pragma unroll
    for (int nt = 0; nt < 8; ++nt) {
      v16h b = load_b_f16(Wtf1, nt * 16 + lm, 256, k0, hi);
      acc[nt] = wmma_16x16x32(a, b, acc[nt]);
    }
  }
#pragma unroll
  for (int nt = 0; nt < 8; ++nt) {
    const int n = nt * 16 + lm;
    const float bias = bf1[n];
#pragma unroll
    for (int v = 0; v < 8; ++v) {
      float hv = fmaxf(acc[nt][v] + bias, 0.0f);
      hid[(w * 16 + v + 8 * hi) * HDIM + n] = (_Float16)hv;
    }
  }
  __syncthreads();

  // layer 2: K = 128
  v8f acc2[8];
#pragma unroll
  for (int nt = 0; nt < 8; ++nt) acc2[nt] = {};
  const int arow = w * 16 + lm;
#pragma unroll
  for (int k0 = 0; k0 < HDIM; k0 += 32) {
    v16h a = load_a_lds(hid, arow, HDIM, k0, hi);
#pragma unroll
    for (int nt = 0; nt < 8; ++nt) {
      v16h b = load_b_f16(Wtf2, nt * 16 + lm, HDIM, k0, hi);
      acc2[nt] = wmma_16x16x32(a, b, acc2[nt]);
    }
  }
#pragma unroll
  for (int nt = 0; nt < 8; ++nt) {
    const int n = nt * 16 + lm;
    const float bias = bf2[n];
#pragma unroll
    for (int v = 0; v < 8; ++v)
      fin[(w * 16 + v + 8 * hi) * HDIM + n] = acc2[nt][v] + bias;
  }
  __syncthreads();

  // layer norm: one thread per row for stats, cooperative coalesced writes
  if (tid < 64) {
    float s = 0.0f;
    for (int n = 0; n < HDIM; ++n) s += fin[tid * HDIM + n];
    const float mu = s * (1.0f / HDIM);
    float v = 0.0f;
    for (int n = 0; n < HDIM; ++n) {
      const float d = fin[tid * HDIM + n] - mu;
      v += d * d;
    }
    mu_s[tid] = mu;
    rs_s[tid] = rsqrtf(v * (1.0f / HDIM) + 1e-5f);
  }
  __syncthreads();
  const float g = ln_g[tid], bb = ln_b[tid];
  for (int m = 0; m < 64; ++m)
    out[(row0 + m) * HDIM + tid] =
        (fin[m * HDIM + tid] - mu_s[m]) * rs_s[m] * g + bb;
}

// ---------------------------------------------------------------------------
extern "C" void kernel_launch(void* const* d_in, const int* in_sizes, int n_in,
                              void* d_out, int out_size, void* d_ws,
                              size_t ws_size, hipStream_t stream) {
  const float* ref_locs = (const float*)d_in[0];
  const float* ref_feat = (const float*)d_in[1];
  const float* q_locs   = (const float*)d_in[2];
  const float* q_feat   = (const float*)d_in[3];
  const float* Wq1 = (const float*)d_in[4];
  const float* bq1 = (const float*)d_in[5];
  const float* Wq2 = (const float*)d_in[6];
  const float* bq2 = (const float*)d_in[7];
  const float* Wr1 = (const float*)d_in[8];
  const float* br1 = (const float*)d_in[9];
  const float* Wr2 = (const float*)d_in[10];
  const float* br2 = (const float*)d_in[11];
  const float* Wf1 = (const float*)d_in[12];
  const float* bf1 = (const float*)d_in[13];
  const float* Wf2 = (const float*)d_in[14];
  const float* bf2 = (const float*)d_in[15];
  const float* ln_g = (const float*)d_in[16];
  const float* ln_b = (const float*)d_in[17];

  float* out  = (float*)d_out;                          // (B,NQ,H)
  float* attn = out + (size_t)BATCH * NQ * HDIM;        // (B,NQ,NR)

  // workspace layout
  float* pq       = (float*)d_ws;
  float* pr       = pq + (size_t)BATCH * NQ * HDIM;
  float* attended = pr + (size_t)BATCH * NR * HDIM;
  _Float16* prT  = (_Float16*)(attended + (size_t)BATCH * NQ * HDIM); // (B,H,NR)
  _Float16* Wtq1 = prT + (size_t)BATCH * HDIM * NR;
  _Float16* Wtq2 = Wtq1 + HDIM * HDIM;
  _Float16* Wtr1 = Wtq2 + HDIM * HDIM;
  _Float16* Wtr2 = Wtr1 + HDIM * HDIM;
  _Float16* Wtf1 = Wtr2 + HDIM * HDIM;   // [128][256]
  _Float16* Wtf2 = Wtf1 + 2 * HDIM * HDIM;

  const int e128 = HDIM * HDIM;
  transpose_f16_kernel<<<(e128 + 255) / 256, 256, 0, stream>>>(Wq1, Wtq1, HDIM, HDIM);
  transpose_f16_kernel<<<(e128 + 255) / 256, 256, 0, stream>>>(Wq2, Wtq2, HDIM, HDIM);
  transpose_f16_kernel<<<(e128 + 255) / 256, 256, 0, stream>>>(Wr1, Wtr1, HDIM, HDIM);
  transpose_f16_kernel<<<(e128 + 255) / 256, 256, 0, stream>>>(Wr2, Wtr2, HDIM, HDIM);
  transpose_f16_kernel<<<(2 * e128 + 255) / 256, 256, 0, stream>>>(Wf1, Wtf1, 2 * HDIM, HDIM);
  transpose_f16_kernel<<<(e128 + 255) / 256, 256, 0, stream>>>(Wf2, Wtf2, HDIM, HDIM);

  const int mlp_blocks = (BATCH * NQ) / 64;  // 512
  mlp2_kernel<<<mlp_blocks, 128, 0, stream>>>(q_feat, Wtq1, bq1, Wtq2, bq2, pq);
  mlp2_kernel<<<mlp_blocks, 128, 0, stream>>>(ref_feat, Wtr1, br1, Wtr2, br2, pr);

  convT_kernel<<<BATCH * (NR / 32), 128, 0, stream>>>(pr, prT);

  attn_kernel<<<BATCH * (NQ / 64), 128, 0, stream>>>(q_locs, ref_locs, prT,
                                                     attn, attended);

  final_kernel<<<mlp_blocks, 128, 0, stream>>>(pq, attended, Wtf1, bf1, Wtf2,
                                               bf2, ln_g, ln_b, out);
}
